// PointerNet_7619271983740
// MI455X (gfx1250) — compile-verified, matrix operand order checked by reading, not run on hardware
//
#include <hip/hip_runtime.h>
#include <stdint.h>

// ---------------------------------------------------------------------------
// Types for CDNA5 WMMA (wave32)
// ---------------------------------------------------------------------------
typedef __attribute__((ext_vector_type(16))) __bf16 v16bf;
typedef __attribute__((ext_vector_type(8)))  __bf16 v8bf;
typedef __attribute__((ext_vector_type(8)))  float  v8f;

#define BATCH   512
#define NNODES  100
#define DMODEL  128
#define NHEAD   8
#define HDIM    16
#define NLAYER  3
#define DFF     2048
#define NROWS   (BATCH * NNODES)          // 51200
#define NSTEPS  (NNODES - 1)              // 99
#define LN_EPS  1e-5f
#define FCHUNK  6400                      // FFN M-chunk rows
#define NCHUNK  (NROWS / FCHUNK)          // 8

static __device__ __forceinline__ __bf16 f2bf(float x) { return (__bf16)x; }

// ---------------------------------------------------------------------------
// f32 -> bf16 conversion
// ---------------------------------------------------------------------------
__global__ void k_f32_to_bf16(const float* __restrict__ src, __bf16* __restrict__ dst, int n) {
    int i = blockIdx.x * blockDim.x + threadIdx.x;
    if (i < n) dst[i] = f2bf(src[i]);
}

// ---------------------------------------------------------------------------
// Embedding: h = c @ w_inp.T + b_inp   (K = 2, scalar path)
// ---------------------------------------------------------------------------
__global__ void k_embed(const float* __restrict__ c, const float* __restrict__ w_inp,
                        const float* __restrict__ b_inp,
                        float* __restrict__ h, __bf16* __restrict__ hbf) {
    int i = blockIdx.x * blockDim.x + threadIdx.x;       // over NROWS*DMODEL
    if (i >= NROWS * DMODEL) return;
    int row = i >> 7, d = i & (DMODEL - 1);
    float v = c[row * 2] * w_inp[d * 2] + c[row * 2 + 1] * w_inp[d * 2 + 1] + b_inp[d];
    h[i] = v;
    hbf[i] = f2bf(v);
}

// ---------------------------------------------------------------------------
// bf16 WMMA GEMM:  out[M,N] = act(A[M,K] @ W[N,K]^T + bias[N])
// block = 256 threads = 8 waves; block tile 128(M) x 64(N); wave tile 16(M) x 64(N)
// with 4 accumulators (A fragment reused 4x). grid = (N/64, M/128).
// Requires M%128==0, N%64==0, K%32==0.
//
// B slab (64 rows x 32 K of the [N,K] weight, 4 KB) is shared by all 8 waves:
// staged in LDS via a double-buffered GLOBAL_LOAD_ASYNC_TO_LDS_B128 pipeline
// (one 16B async load per thread per K-step; s_wait_asynccnt rotates buffers),
// then consumed with ds_load_b128 fragment reads.
//
// Fragment layout per CDNA5 ISA §7.12.2 (wave32, 16-bit A: lane m holds
// K[bk..bk+7] and K[16+bk..bk+7], bk = 8*(lane>>4); B symmetric over N rows
// of the [N,K] weight; C/D: VGPR r -> M = 8*(lane>>4)+r, N = lane&15).
// ---------------------------------------------------------------------------
__global__ void k_gemm_bf16(const __bf16* __restrict__ A, const __bf16* __restrict__ W,
                            const float* __restrict__ bias,
                            float* __restrict__ outF, __bf16* __restrict__ outB,
                            int M, int N, int K, int relu) {
    __shared__ __bf16 sB[2][64 * 32];                 // 2 x 4KB double buffer

    const int tid  = threadIdx.x;
    const int lane = tid & 31;
    const int wave = tid >> 5;
    const int m0 = blockIdx.y * 128 + wave * 16;
    const int n0 = blockIdx.x * 64;
    const int l15 = lane & 15;
    const int bk  = (lane >> 4) * 8;

    const __bf16* ap = A + (size_t)(m0 + l15) * K + bk;

    // async-fill thread mapping: thread -> (row n of slab, 16B k-chunk)
    const int fn = tid >> 2;                          // 0..63
    const int fk = (tid & 3) * 8;                     // 0,8,16,24 (elements)
    const __bf16* wfill = W + (size_t)(n0 + fn) * K + fk;
    const int ldst = fn * 32 + fk;                    // element offset in slab

    const int nk = K >> 5;                            // K/32 steps

    // helper: issue one async 16B global->LDS load for slab k-step `ks` into buf
    auto fill = [&](int buf, int ks) {
        uint64_t ga = (uint64_t)(uintptr_t)(wfill + ks * 32);
        unsigned  la = (unsigned)(uintptr_t)(&sB[buf][ldst]);  // low 32 bits = LDS byte offset
        asm volatile("global_load_async_to_lds_b128 %0, %1, off"
                     :: "v"(la), "v"(ga) : "memory");
    };

    v8f acc0 = {}, acc1 = {}, acc2 = {}, acc3 = {};

    fill(0, 0);
    for (int ks = 0; ks < nk; ++ks) {
        const int buf = ks & 1;
        const bool more = (ks + 1) < nk;
        if (more) fill(buf ^ 1, ks + 1);
        if (more) asm volatile("s_wait_asynccnt 0x1" ::: "memory");
        else      asm volatile("s_wait_asynccnt 0x0" ::: "memory");
        __syncthreads();                               // slab `buf` visible to all waves

        // A fragment (two contiguous 16B global loads per lane)
        v8bf alo = *(const v8bf*)(ap + ks * 32);
        v8bf ahi = *(const v8bf*)(ap + ks * 32 + 16);
        v16bf a = __builtin_shufflevector(alo, ahi, 0,1,2,3,4,5,6,7,8,9,10,11,12,13,14,15);

        // 4 B fragments from LDS, 4 WMMAs
        #pragma unroll
        for (int nt = 0; nt < 4; ++nt) {
            const __bf16* bp = &sB[buf][(nt * 16 + l15) * 32 + bk];
            v8bf blo = *(const v8bf*)(bp);
            v8bf bhi = *(const v8bf*)(bp + 16);
            v16bf b = __builtin_shufflevector(blo, bhi, 0,1,2,3,4,5,6,7,8,9,10,11,12,13,14,15);
            if      (nt == 0) acc0 = __builtin_amdgcn_wmma_f32_16x16x32_bf16(false, a, false, b, (short)0, acc0, false, false);
            else if (nt == 1) acc1 = __builtin_amdgcn_wmma_f32_16x16x32_bf16(false, a, false, b, (short)0, acc1, false, false);
            else if (nt == 2) acc2 = __builtin_amdgcn_wmma_f32_16x16x32_bf16(false, a, false, b, (short)0, acc2, false, false);
            else              acc3 = __builtin_amdgcn_wmma_f32_16x16x32_bf16(false, a, false, b, (short)0, acc3, false, false);
        }
        __syncthreads();                               // all waves done before slab overwrite
    }

    // epilogue: C/D layout -> VGPR r holds (M = mbase+r, N = n0 + nt*16 + (lane&15))
    const int mbase = m0 + (lane >> 4) * 8;
    #pragma unroll
    for (int nt = 0; nt < 4; ++nt) {
        const int nc = n0 + nt * 16 + l15;
        const float bv = bias ? bias[nc] : 0.0f;
        const v8f* accp = (nt == 0) ? &acc0 : (nt == 1) ? &acc1 : (nt == 2) ? &acc2 : &acc3;
        #pragma unroll
        for (int r = 0; r < 8; ++r) {
            float v = (*accp)[r] + bv;
            if (relu) v = fmaxf(v, 0.0f);
            size_t o = (size_t)(mbase + r) * N + nc;
            if (outF) outF[o] = v;
            if (outB) outB[o] = f2bf(v);
        }
    }
    (void)M;
}

// ---------------------------------------------------------------------------
// Attention core: per (b, head) block; softmax(q k^T / sqrt(hd)) v
// qkv: bf16 [NROWS, 3*DMODEL]; out: bf16 [NROWS, DMODEL]
// ---------------------------------------------------------------------------
__global__ void k_attention(const __bf16* __restrict__ qkv, __bf16* __restrict__ out) {
    __shared__ float ks[NNODES * HDIM];
    __shared__ float vs[NNODES * HDIM];
    const int b = blockIdx.x / NHEAD;
    const int h = blockIdx.x % NHEAD;
    const int tid = threadIdx.x;
    const size_t base = (size_t)b * NNODES;

    for (int idx = tid; idx < NNODES * HDIM; idx += blockDim.x) {
        int n = idx / HDIM, d = idx % HDIM;
        size_t r = (base + n) * (3 * DMODEL) + h * HDIM + d;
        ks[idx] = (float)qkv[r + DMODEL];
        vs[idx] = (float)qkv[r + 2 * DMODEL];
    }
    __syncthreads();

    if (tid < NNODES) {
        float q[HDIM];
        size_t qr = (base + tid) * (3 * DMODEL) + h * HDIM;
        #pragma unroll
        for (int d = 0; d < HDIM; ++d) q[d] = (float)qkv[qr + d];

        const float scale = 0.25f;   // hd^-0.5, hd=16
        float mx = -3.402823466e38f;
        for (int j = 0; j < NNODES; ++j) {
            float s = 0.f;
            #pragma unroll
            for (int d = 0; d < HDIM; ++d) s += q[d] * ks[j * HDIM + d];
            s *= scale;
            mx = fmaxf(mx, s);
        }
        float accv[HDIM];
        #pragma unroll
        for (int d = 0; d < HDIM; ++d) accv[d] = 0.f;
        float sum = 0.f;
        for (int j = 0; j < NNODES; ++j) {
            float s = 0.f;
            #pragma unroll
            for (int d = 0; d < HDIM; ++d) s += q[d] * ks[j * HDIM + d];
            float p = expf(s * scale - mx);
            sum += p;
            #pragma unroll
            for (int d = 0; d < HDIM; ++d) accv[d] += p * vs[j * HDIM + d];
        }
        float inv = 1.0f / sum;
        size_t orow = (base + tid) * DMODEL + h * HDIM;
        #pragma unroll
        for (int d = 0; d < HDIM; ++d) out[orow + d] = f2bf(accv[d] * inv);
    }
}

// ---------------------------------------------------------------------------
// Residual + LayerNorm (in place on h), also emits bf16 copy.
// One wave per row (wave32, 4 elems/lane). block = 256 = 8 rows.
// ---------------------------------------------------------------------------
__global__ void k_resid_ln(float* __restrict__ h, const float* __restrict__ delta,
                           const float* __restrict__ g, const float* __restrict__ be,
                           __bf16* __restrict__ hbf) {
    const int lane = threadIdx.x & 31;
    const int wave = threadIdx.x >> 5;
    const int row = blockIdx.x * 8 + wave;
    const size_t off = (size_t)row * DMODEL + lane * 4;

    float x[4];
    #pragma unroll
    for (int j = 0; j < 4; ++j) x[j] = h[off + j] + delta[off + j];

    float s = x[0] + x[1] + x[2] + x[3];
    #pragma unroll
    for (int o = 16; o > 0; o >>= 1) s += __shfl_xor(s, o);
    const float mean = s * (1.0f / DMODEL);

    float vsum = 0.f;
    #pragma unroll
    for (int j = 0; j < 4; ++j) { float d = x[j] - mean; vsum += d * d; }
    #pragma unroll
    for (int o = 16; o > 0; o >>= 1) vsum += __shfl_xor(vsum, o);
    const float rstd = rsqrtf(vsum * (1.0f / DMODEL) + LN_EPS);

    #pragma unroll
    for (int j = 0; j < 4; ++j) {
        int col = lane * 4 + j;
        float y = (x[j] - mean) * rstd * g[col] + be[col];
        h[off + j] = y;
        hbf[off + j] = f2bf(y);
    }
}

// ---------------------------------------------------------------------------
// Decode init: visited mask (depot pre-visited) + current-node state
// ---------------------------------------------------------------------------
__global__ void k_decode_init(const int* __restrict__ start, unsigned char* __restrict__ visited,
                              int* __restrict__ node_state) {
    int i = blockIdx.x * blockDim.x + threadIdx.x;
    if (i < NROWS) visited[i] = (i % NNODES == 0) ? 1 : 0;
    if (i < BATCH) node_state[i] = start[i];
}

// ---------------------------------------------------------------------------
// One greedy decode step: q = e[b,node] @ wq.T + bq; logits = q . kproj / sqrt(D);
// mask visited (incl. current); record tour + logits; argmax -> next node.
// block = 128 threads, grid = BATCH.
// ---------------------------------------------------------------------------
__global__ void k_decode_step(const float* __restrict__ e, const float* __restrict__ kproj,
                              const float* __restrict__ wq, const float* __restrict__ bq,
                              unsigned char* __restrict__ visited, int* __restrict__ node_state,
                              float* __restrict__ out, int t) {
    __shared__ float erow[DMODEL];
    __shared__ float qv[DMODEL];
    __shared__ float rv[128];
    __shared__ int   ri[128];

    const int b = blockIdx.x;
    const int j = threadIdx.x;
    const int node = node_state[b];

    if (j == 0) visited[b * NNODES + node] = 1;   // mask current node
    erow[j] = e[((size_t)b * NNODES + node) * DMODEL + j];
    __syncthreads();

    float q = bq[j];
    const float* wr = wq + (size_t)j * DMODEL;
    #pragma unroll 8
    for (int k = 0; k < DMODEL; ++k) q += erow[k] * wr[k];
    qv[j] = q;
    __syncthreads();

    const float scale = 0.0883883476483184f;      // D^-0.5, D=128
    float val = -__builtin_inff();
    if (j < NNODES) {
        const float* kr = kproj + ((size_t)b * NNODES + j) * DMODEL;
        float s = 0.f;
        #pragma unroll 8
        for (int k = 0; k < DMODEL; ++k) s += qv[k] * kr[k];
        s *= scale;
        if (visited[b * NNODES + j]) s = -3.4028234663852886e38f;  // finfo(f32).min
        val = s;
        out[(size_t)BATCH * NSTEPS + ((size_t)b * NSTEPS + t) * NNODES + j] = s;
    }
    rv[j] = val; ri[j] = j;
    __syncthreads();
    for (int st = 64; st > 0; st >>= 1) {
        if (j < st) {
            if (rv[j + st] > rv[j]) { rv[j] = rv[j + st]; ri[j] = ri[j + st]; }
        }
        __syncthreads();
    }
    if (j == 0) {
        node_state[b] = ri[0];
        out[(size_t)b * NSTEPS + t] = (float)node;   // tour records current node
    }
}

// ---------------------------------------------------------------------------
// Host-side orchestration
// ---------------------------------------------------------------------------
extern "C" void kernel_launch(void* const* d_in, const int* in_sizes, int n_in,
                              void* d_out, int out_size, void* d_ws, size_t ws_size,
                              hipStream_t stream) {
    (void)in_sizes; (void)n_in; (void)out_size; (void)ws_size;

    const float* c      = (const float*)d_in[0];
    const int*   start  = (const int*)  d_in[1];
    const float* w_inp  = (const float*)d_in[2];
    const float* b_inp  = (const float*)d_in[3];
    const float* wqkv   = (const float*)d_in[4];
    const float* bqkv   = (const float*)d_in[5];
    const float* wo     = (const float*)d_in[6];
    const float* bo     = (const float*)d_in[7];
    const float* w1     = (const float*)d_in[8];
    const float* b1     = (const float*)d_in[9];
    const float* w2     = (const float*)d_in[10];
    const float* b2     = (const float*)d_in[11];
    const float* g1     = (const float*)d_in[12];
    const float* be1    = (const float*)d_in[13];
    const float* g2     = (const float*)d_in[14];
    const float* be2    = (const float*)d_in[15];
    const float* wq     = (const float*)d_in[16];
    const float* bq     = (const float*)d_in[17];
    const float* wk     = (const float*)d_in[18];
    const float* bk     = (const float*)d_in[19];

    // ---- workspace layout (bytes, 256-aligned) ----
    auto align256 = [](size_t x) { return (x + 255) & ~(size_t)255; };
    char* ws = (char*)d_ws;
    size_t off = 0;
    float* h      = (float*)(ws + off); off = align256(off + (size_t)NROWS * DMODEL * 4);
    float* proj   = (float*)(ws + off); off = align256(off + (size_t)NROWS * DMODEL * 4);
    float* kproj  = (float*)(ws + off); off = align256(off + (size_t)NROWS * DMODEL * 4);
    __bf16* hbf    = (__bf16*)(ws + off); off = align256(off + (size_t)NROWS * DMODEL * 2);
    __bf16* attnbf = (__bf16*)(ws + off); off = align256(off + (size_t)NROWS * DMODEL * 2);
    __bf16* qkvbf  = (__bf16*)(ws + off); off = align256(off + (size_t)NROWS * 3 * DMODEL * 2);
    __bf16* ffbf   = (__bf16*)(ws + off); off = align256(off + (size_t)FCHUNK * DFF * 2);
    __bf16* wqkvbf = (__bf16*)(ws + off); off = align256(off + (size_t)NLAYER * 3 * DMODEL * DMODEL * 2);
    __bf16* wobf   = (__bf16*)(ws + off); off = align256(off + (size_t)NLAYER * DMODEL * DMODEL * 2);
    __bf16* w1bf   = (__bf16*)(ws + off); off = align256(off + (size_t)NLAYER * DFF * DMODEL * 2);
    __bf16* w2bf   = (__bf16*)(ws + off); off = align256(off + (size_t)NLAYER * DMODEL * DFF * 2);
    __bf16* wkbf   = (__bf16*)(ws + off); off = align256(off + (size_t)DMODEL * DMODEL * 2);
    unsigned char* visited = (unsigned char*)(ws + off); off = align256(off + NROWS);
    int* node_state = (int*)(ws + off); off = align256(off + BATCH * 4);

    auto conv = [&](const float* s, __bf16* d, int n) {
        k_f32_to_bf16<<<(n + 255) / 256, 256, 0, stream>>>(s, d, n);
    };
    conv(wqkv, wqkvbf, NLAYER * 3 * DMODEL * DMODEL);
    conv(wo,   wobf,   NLAYER * DMODEL * DMODEL);
    conv(w1,   w1bf,   NLAYER * DFF * DMODEL);
    conv(w2,   w2bf,   NLAYER * DMODEL * DFF);
    conv(wk,   wkbf,   DMODEL * DMODEL);

    // ---- embed ----
    k_embed<<<(NROWS * DMODEL + 255) / 256, 256, 0, stream>>>(c, w_inp, b_inp, h, hbf);

    // ---- encoder layers ----
    for (int l = 0; l < NLAYER; ++l) {
        const __bf16* wqkv_l = wqkvbf + (size_t)l * 3 * DMODEL * DMODEL;
        const __bf16* wo_l   = wobf   + (size_t)l * DMODEL * DMODEL;
        const __bf16* w1_l   = w1bf   + (size_t)l * DFF * DMODEL;
        const __bf16* w2_l   = w2bf   + (size_t)l * DMODEL * DFF;

        // qkv = h @ wqkv.T + bqkv  -> bf16
        k_gemm_bf16<<<dim3(3 * DMODEL / 64, NROWS / 128), 256, 0, stream>>>(
            hbf, wqkv_l, bqkv + l * 3 * DMODEL, nullptr, qkvbf,
            NROWS, 3 * DMODEL, DMODEL, 0);

        // attention -> bf16
        k_attention<<<BATCH * NHEAD, 128, 0, stream>>>(qkvbf, attnbf);

        // out proj -> f32
        k_gemm_bf16<<<dim3(DMODEL / 64, NROWS / 128), 256, 0, stream>>>(
            attnbf, wo_l, bo + l * DMODEL, proj, nullptr,
            NROWS, DMODEL, DMODEL, 0);

        // h = LN(h + attn_proj)
        k_resid_ln<<<NROWS / 8, 256, 0, stream>>>(h, proj, g1 + l * DMODEL, be1 + l * DMODEL, hbf);

        // FFN, chunked over M
        for (int ch = 0; ch < NCHUNK; ++ch) {
            size_t r0 = (size_t)ch * FCHUNK;
            k_gemm_bf16<<<dim3(DFF / 64, FCHUNK / 128), 256, 0, stream>>>(
                hbf + r0 * DMODEL, w1_l, b1 + l * DFF, nullptr, ffbf,
                FCHUNK, DFF, DMODEL, 1);
            k_gemm_bf16<<<dim3(DMODEL / 64, FCHUNK / 128), 256, 0, stream>>>(
                ffbf, w2_l, b2 + l * DMODEL, proj + r0 * DMODEL, nullptr,
                FCHUNK, DMODEL, DFF, 0);
        }

        // h = LN(h + ffn)
        k_resid_ln<<<NROWS / 8, 256, 0, stream>>>(h, proj, g2 + l * DMODEL, be2 + l * DMODEL, hbf);
    }

    // ---- kproj = e @ wk.T + bk (loop-invariant) ----
    k_gemm_bf16<<<dim3(DMODEL / 64, NROWS / 128), 256, 0, stream>>>(
        hbf, wkbf, bk, kproj, nullptr, NROWS, DMODEL, DMODEL, 0);

    // ---- greedy decode ----
    k_decode_init<<<(NROWS + 255) / 256, 256, 0, stream>>>(start, visited, node_state);
    for (int t = 0; t < NSTEPS; ++t) {
        k_decode_step<<<BATCH, 128, 0, stream>>>(h, kproj, wq, bq, visited, node_state,
                                                 (float*)d_out, t);
    }
}
